// CausalSelfAttention_58634893525223
// MI455X (gfx1250) — compile-verified, matrix operand order checked by reading, not run on hardware
//
#include <hip/hip_runtime.h>
#include <math.h>

// Problem constants (match reference setup_inputs)
#define BB   2
#define TT   2048
#define DD   1024
#define HH   16
#define DH   64
#define N3   3072                      // 3*D
#define MM   (BB*TT)                   // 4096 rows of x
#define SBUF (BB*HH*TT*DH)             // 4,194,304 elems per q/k/v buffer

typedef __attribute__((ext_vector_type(16))) _Float16 v16h;
typedef __attribute__((ext_vector_type(8)))  float    v8f;
typedef __attribute__((ext_vector_type(4)))  int      v4i;

#if defined(__has_builtin)
#if __has_builtin(__builtin_amdgcn_global_load_async_to_lds_b128) && \
    __has_builtin(__builtin_amdgcn_s_wait_asynccnt)
#define HAS_ASYNC_LDS 1
typedef __attribute__((address_space(1))) v4i glob_v4i;   // global int4
typedef __attribute__((address_space(3))) v4i lds_v4i;    // LDS int4
#endif
#endif

// f16 workspace layout (element offsets)
#define OFF_XH   ((size_t)0)                       // x as f16         (MM*DD)
#define OFF_WH   (OFF_XH  + (size_t)MM*DD)         // W_qkv as f16     (DD*N3)
#define OFF_WOH  (OFF_WH  + (size_t)DD*N3)         // W_out as f16     (DD*DD)
#define OFF_Q    (OFF_WOH + (size_t)DD*DD)         // q (B,H,T,Dh) f16
#define OFF_K    (OFF_Q   + (size_t)SBUF)
#define OFF_V    (OFF_K   + (size_t)SBUF)
#define OFF_CTX  (OFF_V   + (size_t)SBUF)          // ctx (B,T,H*Dh) f16

// ---------------------------------------------------------------------------
// Kernel 0: f32 -> f16 conversion (4 elements / thread)
// ---------------------------------------------------------------------------
__global__ __launch_bounds__(256) void cvt_f16_kernel(
    const float* __restrict__ src, _Float16* __restrict__ dst) {
  const size_t i = ((size_t)blockIdx.x * 256 + threadIdx.x) * 4;
#pragma unroll
  for (int e = 0; e < 4; ++e) dst[i + e] = (_Float16)src[i + e];
}

// ===========================================================================
// Register-blocked f16 GEMM core: one wave owns a 64x64 fp32 tile of
// C = A(f16,MxK) @ B(f16,KxN) via 4x4 v_wmma_f32_16x16x32_f16 accumulators.
// A fragments: direct global f16 loads (16B contiguous per lane).
// B panel (32x64 f16 = 4KB) is double-buffered in LDS and filled with
// global_load_async_to_lds_b128 (2 instrs/thread/panel), pipelined with
// s_wait_asynccnt 2. Block = 4 waves stacked in M (256x64 block tile).
// ===========================================================================
__device__ __forceinline__ void gemm_block_256x64_h(
    const _Float16* __restrict__ A, const _Float16* __restrict__ Bm,
    int lda, int ldb, int blk_m0, int n0, int K,
    _Float16* __restrict__ ldsB /* [2][32*64] */, v8f acc[4][4]) {
  const int tid   = threadIdx.x;
  const int wave  = tid >> 5;
  const int lane  = tid & 31;
  const int lrow  = lane & 15;
  const int half8 = (lane >> 4) * 8;
  const int m0    = blk_m0 + wave * 64;

  const _Float16* arow[4];
#pragma unroll
  for (int i = 0; i < 4; ++i)
    arow[i] = A + (size_t)(m0 + i * 16 + lrow) * lda;

  // B panel staging: thread t covers row (t>>2), cols (t&3)*16 + {0..7, 8..15}
  const int brow  = tid >> 2;
  const int bcol0 = (tid & 3) * 16;

  auto issue = [&](int k0, int buf) {
    const _Float16* src = Bm + (size_t)(k0 + brow) * ldb + n0 + bcol0;
    _Float16* dst = ldsB + buf * 2048 + brow * 64 + bcol0;
#if HAS_ASYNC_LDS
    __builtin_amdgcn_global_load_async_to_lds_b128(
        (glob_v4i*)src, (lds_v4i*)dst, 0, 0);
    __builtin_amdgcn_global_load_async_to_lds_b128(
        (glob_v4i*)(src + 8), (lds_v4i*)(dst + 8), 0, 0);
#else
#pragma unroll
    for (int e = 0; e < 16; ++e) dst[e] = src[e];
#endif
  };

  auto compute_panel = [&](int k0, int buf) {
    if (k0 + 32 < K) {                       // prefetch next A segments to L2
#pragma unroll
      for (int i = 0; i < 4; ++i)
        __builtin_prefetch(arow[i] + k0 + 32, 0, 1);
    }
    v16h af[4], bf[4];
    const _Float16* lp = ldsB + buf * 2048;
#pragma unroll
    for (int i = 0; i < 4; ++i) {
#pragma unroll
      for (int e = 0; e < 8; ++e) {
        af[i][e]     = arow[i][k0 + half8 + e];
        af[i][8 + e] = arow[i][k0 + 16 + half8 + e];
      }
    }
#pragma unroll
    for (int j = 0; j < 4; ++j) {
      const int col = j * 16 + lrow;
#pragma unroll
      for (int e = 0; e < 8; ++e) {
        bf[j][e]     = lp[(half8 + e) * 64 + col];
        bf[j][8 + e] = lp[(16 + half8 + e) * 64 + col];
      }
    }
#pragma unroll
    for (int i = 0; i < 4; ++i)
#pragma unroll
      for (int j = 0; j < 4; ++j)
        acc[i][j] = __builtin_amdgcn_wmma_f32_16x16x32_f16(
            false, af[i], false, bf[j], (short)0, acc[i][j], false, false);
  };

  issue(0, 0);
  int cur = 0, k0 = 0;
  for (; k0 + 32 < K; k0 += 32) {
    issue(k0 + 32, cur ^ 1);                 // overlap next panel fetch
#if HAS_ASYNC_LDS
    __builtin_amdgcn_s_wait_asynccnt(2);     // current panel resident
#endif
    __syncthreads();
    compute_panel(k0, cur);
    __syncthreads();                         // done reading before next write
    cur ^= 1;
  }
#if HAS_ASYNC_LDS
  __builtin_amdgcn_s_wait_asynccnt(0);
#endif
  __syncthreads();
  compute_panel(k0, cur);
}

// ---------------------------------------------------------------------------
// Kernel 1: QKV = xh @ Wh, scattered as f16 into (which, B, H, T, Dh)
// ---------------------------------------------------------------------------
__global__ __launch_bounds__(128) void qkv_gemm_kernel(_Float16* __restrict__ hws) {
  __shared__ _Float16 ldsB[2 * 32 * 64];
  const int NB = N3 / 64;                         // 48
  const int bm = blockIdx.x / NB;
  const int bn = blockIdx.x % NB;
  const int blk_m0 = bm * 256;
  const int n0 = bn * 64;

  v8f acc[4][4];
#pragma unroll
  for (int i = 0; i < 4; ++i)
#pragma unroll
    for (int j = 0; j < 4; ++j) acc[i][j] = (v8f){};

  gemm_block_256x64_h(hws + OFF_XH, hws + OFF_WH, DD, N3, blk_m0, n0, DD,
                      ldsB, acc);

  const int wave = threadIdx.x >> 5;
  const int lane = threadIdx.x & 31;
  const int lrow = lane & 15;
  const int hi8  = (lane >> 4) * 8;
  const int m0   = blk_m0 + wave * 64;
#pragma unroll
  for (int j = 0; j < 4; ++j) {
    const int n     = n0 + j * 16 + lrow;
    const int which = n >> 10;                    // 0=q 1=k 2=v
    const int c     = n & 1023;
    const int h     = c >> 6;
    const int dh    = c & 63;
    _Float16* dstbuf = hws + OFF_Q + (size_t)which * SBUF;
#pragma unroll
    for (int i = 0; i < 4; ++i) {
#pragma unroll
      for (int r = 0; r < 8; ++r) {
        const int m = m0 + i * 16 + r + hi8;
        const int b = m >> 11;                    // /T
        const int t = m & (TT - 1);
        dstbuf[((size_t)(b * HH + h) * TT + t) * DH + dh] = (_Float16)acc[i][j][r];
      }
    }
  }
}

// ---------------------------------------------------------------------------
// Kernel 2: in-place RoPE on q and k f16 buffers (layout B,H,T,Dh)
// ---------------------------------------------------------------------------
__global__ __launch_bounds__(256) void rope_kernel(_Float16* __restrict__ hws) {
  const int gid = blockIdx.x * 256 + threadIdx.x;   // 2 * B*H*T*32 threads
  const int perbuf = BB * HH * TT * 32;
  const int sel = gid / perbuf;                      // 0=q, 1=k
  const int rem = gid % perbuf;
  const int row = rem >> 5;                          // (b,h,t) flat row
  const int d   = rem & 31;
  const int t   = row & (TT - 1);
  _Float16* base = hws + OFF_Q + (size_t)sel * SBUF + (size_t)row * DH;

  const float inv_freq = __powf(10000.0f, -(float)d * (1.0f / 32.0f));
  float s, cst;
  __sincosf((float)t * inv_freq, &s, &cst);
  const float u1 = (float)base[d];
  const float u2 = (float)base[d + 32];
  base[d]      = (_Float16)(u1 * cst - u2 * s);
  base[d + 32] = (_Float16)(u2 * cst + u1 * s);
}

// ---------------------------------------------------------------------------
// Kernel 3: flash-attention, all-f16 operands. One wave per (b,h,16-q tile).
// V blocks stream through LDS via async DMA overlapped with S + softmax.
// ---------------------------------------------------------------------------
__global__ __launch_bounds__(128) void attn_kernel(_Float16* __restrict__ hws) {
  __shared__ _Float16 ldsP[4][16 * 32];             // per-wave P tile (1KB)
#if HAS_ASYNC_LDS
  __shared__ _Float16 ldsV[4][32 * 64];             // per-wave V block (4KB)
#endif

  const int wave = threadIdx.x >> 5;
  const int job  = blockIdx.x * 4 + wave;           // 32 bh * 128 qtiles
  const int bh   = job >> 7;
  const int qt   = job & 127;
  const int t0   = qt * 16;
  const int b    = bh >> 4;
  const int h    = bh & 15;

  const _Float16* Q = hws + OFF_Q + (size_t)bh * TT * DH;
  const _Float16* K = hws + OFF_K + (size_t)bh * TT * DH;
  const _Float16* V = hws + OFF_V + (size_t)bh * TT * DH;
  _Float16* ctx = hws + OFF_CTX;

  const int lane  = threadIdx.x & 31;
  const int lrow  = lane & 15;
  const int hi    = lane >> 4;
  const int half8 = hi * 8;
  _Float16* myP = ldsP[wave];
#if HAS_ASYNC_LDS
  _Float16* myV = ldsV[wave];
#endif

  // Q tile as A-operand for the two 32-wide K-dim steps (no cvt needed)
  v16h qa0, qa1;
  {
    const _Float16* qr = Q + (size_t)(t0 + lrow) * DH;
#pragma unroll
    for (int e = 0; e < 8; ++e) {
      qa0[e]     = qr[half8 + e];
      qa0[8 + e] = qr[16 + half8 + e];
      qa1[e]     = qr[32 + half8 + e];
      qa1[8 + e] = qr[48 + half8 + e];
    }
  }

  v8f o0 = {}, o1 = {}, o2 = {}, o3 = {};
  float rm[8], rl[8];
#pragma unroll
  for (int r = 0; r < 8; ++r) { rm[r] = -__builtin_inff(); rl[r] = 0.0f; }

  const int kbmax = (t0 + 15) >> 5;                 // key blocks of 32
  for (int kb = 0; kb <= kbmax; ++kb) {
    const _Float16* vb = V + (size_t)(kb * 32) * DH;

#if HAS_ASYNC_LDS
    // async-stage the 32x64 f16 V block (4KB): 256 b128 chunks, 8 per lane
#pragma unroll
    for (int i = 0; i < 8; ++i) {
      const int idx = i * 32 + lane;                // b128 (8 x f16) index
      __builtin_amdgcn_global_load_async_to_lds_b128(
          (glob_v4i*)(vb + idx * 8), (lds_v4i*)(myV + idx * 8), 0, 0);
    }
#endif

    // ---- S = Q @ K^T for 32 keys: two 16x16 C tiles ----
    v8f s0 = {}, s1 = {};
    {
      v16h b0, b1;
      const _Float16* kr0 = K + (size_t)(kb * 32 + lrow) * DH;
      const _Float16* kr1 = K + (size_t)(kb * 32 + 16 + lrow) * DH;
#pragma unroll
      for (int e = 0; e < 8; ++e) {                 // K-dims 0..31
        b0[e] = kr0[half8 + e];      b0[8 + e] = kr0[16 + half8 + e];
        b1[e] = kr1[half8 + e];      b1[8 + e] = kr1[16 + half8 + e];
      }
      s0 = __builtin_amdgcn_wmma_f32_16x16x32_f16(false, qa0, false, b0, (short)0, s0, false, false);
      s1 = __builtin_amdgcn_wmma_f32_16x16x32_f16(false, qa0, false, b1, (short)0, s1, false, false);
#pragma unroll
      for (int e = 0; e < 8; ++e) {                 // K-dims 32..63
        b0[e] = kr0[32 + half8 + e]; b0[8 + e] = kr0[48 + half8 + e];
        b1[e] = kr1[32 + half8 + e]; b1[8 + e] = kr1[48 + half8 + e];
      }
      s0 = __builtin_amdgcn_wmma_f32_16x16x32_f16(false, qa1, false, b0, (short)0, s0, false, false);
      s1 = __builtin_amdgcn_wmma_f32_16x16x32_f16(false, qa1, false, b1, (short)0, s1, false, false);
    }

    // ---- online softmax; 1/sqrt(Dh)=0.125 applied to scores here ----
    const int key0 = kb * 32 + lrow;
    const int key1 = kb * 32 + 16 + lrow;
    float p0[8], p1[8], corr[8];
#pragma unroll
    for (int r = 0; r < 8; ++r) {
      const int trow = t0 + r + hi * 8;
      float a = (key0 > trow) ? -__builtin_inff() : s0[r] * 0.125f;
      float c = (key1 > trow) ? -__builtin_inff() : s1[r] * 0.125f;
      float m = fmaxf(a, c);
      m = fmaxf(m, __shfl_xor(m, 1, 32));
      m = fmaxf(m, __shfl_xor(m, 2, 32));
      m = fmaxf(m, __shfl_xor(m, 4, 32));
      m = fmaxf(m, __shfl_xor(m, 8, 32));           // row max within 16-lane half
      const float mnew = fmaxf(rm[r], m);
      corr[r] = __expf(rm[r] - mnew);
      rm[r] = mnew;
      p0[r] = __expf(a - mnew);                     // exp(-inf)=0 handles mask
      p1[r] = __expf(c - mnew);
      float sum = p0[r] + p1[r];
      sum += __shfl_xor(sum, 1, 32);
      sum += __shfl_xor(sum, 2, 32);
      sum += __shfl_xor(sum, 4, 32);
      sum += __shfl_xor(sum, 8, 32);
      rl[r] = rl[r] * corr[r] + sum;
      o0[r] *= corr[r]; o1[r] *= corr[r]; o2[r] *= corr[r]; o3[r] *= corr[r];
      // C-layout -> LDS (row-major 16x32 P tile, stored as f16)
      myP[(r + hi * 8) * 32 + lrow]      = (_Float16)p0[r];
      myP[(r + hi * 8) * 32 + 16 + lrow] = (_Float16)p1[r];
    }

    // ---- reload P in A-operand layout (LDS in-order within wave) ----
    v16h pa;
#pragma unroll
    for (int e = 0; e < 8; ++e) {
      pa[e]     = myP[lrow * 32 + half8 + e];
      pa[8 + e] = myP[lrow * 32 + 16 + half8 + e];
    }

#if HAS_ASYNC_LDS
    __builtin_amdgcn_s_wait_asynccnt(0);            // V block resident in LDS
#endif

    // ---- O += P @ V (K=32 keys, 4 column chunks of 16) ----
#pragma unroll
    for (int nt = 0; nt < 4; ++nt) {
      v16h bv;
      const int col = nt * 16 + lrow;
#pragma unroll
      for (int e = 0; e < 8; ++e) {
#if HAS_ASYNC_LDS
        bv[e]     = myV[(half8 + e) * DH + col];
        bv[8 + e] = myV[(16 + half8 + e) * DH + col];
#else
        bv[e]     = vb[(size_t)(half8 + e) * DH + col];
        bv[8 + e] = vb[(size_t)(16 + half8 + e) * DH + col];
#endif
      }
      v8f* op = (nt == 0) ? &o0 : (nt == 1) ? &o1 : (nt == 2) ? &o2 : &o3;
      *op = __builtin_amdgcn_wmma_f32_16x16x32_f16(false, pa, false, bv, (short)0, *op, false, false);
    }
  }

  // ---- normalize and store f16 context (B, T, H*Dh) ----
#pragma unroll
  for (int r = 0; r < 8; ++r) {
    const float inv = 1.0f / rl[r];
    const size_t row = (size_t)(b * TT + t0 + r + hi * 8) * DD + h * DH;
    ctx[row + 0 * 16 + lrow] = (_Float16)(o0[r] * inv);
    ctx[row + 1 * 16 + lrow] = (_Float16)(o1[r] * inv);
    ctx[row + 2 * 16 + lrow] = (_Float16)(o2[r] * inv);
    ctx[row + 3 * 16 + lrow] = (_Float16)(o3[r] * inv);
  }
}

// ---------------------------------------------------------------------------
// Kernel 4: out(f32) = ctx(f16) @ W_out(f16)   (4096 x 1024 x 1024)
// ---------------------------------------------------------------------------
__global__ __launch_bounds__(128) void outproj_kernel(
    const _Float16* __restrict__ hws, float* __restrict__ out) {
  __shared__ _Float16 ldsB[2 * 32 * 64];
  const int NB = DD / 64;                         // 16
  const int bm = blockIdx.x / NB;
  const int bn = blockIdx.x % NB;
  const int blk_m0 = bm * 256;
  const int n0 = bn * 64;

  v8f acc[4][4];
#pragma unroll
  for (int i = 0; i < 4; ++i)
#pragma unroll
    for (int j = 0; j < 4; ++j) acc[i][j] = (v8f){};

  gemm_block_256x64_h(hws + OFF_CTX, hws + OFF_WOH, DD, DD, blk_m0, n0, DD,
                      ldsB, acc);

  const int wave = threadIdx.x >> 5;
  const int lane = threadIdx.x & 31;
  const int lrow = lane & 15;
  const int hi8  = (lane >> 4) * 8;
  const int m0   = blk_m0 + wave * 64;
#pragma unroll
  for (int i = 0; i < 4; ++i)
#pragma unroll
    for (int j = 0; j < 4; ++j)
#pragma unroll
      for (int r = 0; r < 8; ++r)
        out[(size_t)(m0 + i * 16 + r + hi8) * DD + n0 + j * 16 + lrow] =
            acc[i][j][r];
}

// ---------------------------------------------------------------------------
extern "C" void kernel_launch(void* const* d_in, const int* in_sizes, int n_in,
                              void* d_out, int out_size, void* d_ws, size_t ws_size,
                              hipStream_t stream) {
  const float* x    = (const float*)d_in[0];
  const float* Wqkv = (const float*)d_in[1];
  const float* Wout = (const float*)d_in[2];
  float* out = (float*)d_out;
  _Float16* hws = (_Float16*)d_ws;         // all-f16 workspace

  // 0) one-time f32 -> f16 conversion of inputs (4 elems/thread)
  cvt_f16_kernel<<<(MM * DD) / 1024, 256, 0, stream>>>(x,    hws + OFF_XH);
  cvt_f16_kernel<<<((size_t)DD * N3) / 1024, 256, 0, stream>>>(Wqkv, hws + OFF_WH);
  cvt_f16_kernel<<<((size_t)DD * DD) / 1024, 256, 0, stream>>>(Wout, hws + OFF_WOH);

  // 1) QKV projection: (4096/256) * (3072/64) = 16*48 = 768 blocks
  qkv_gemm_kernel<<<768, 128, 0, stream>>>(hws);
  // 2) RoPE on q,k: 2 * B*H*T*32 = 4,194,304 threads
  rope_kernel<<<(2 * BB * HH * TT * 32) / 256, 256, 0, stream>>>(hws);
  // 3) attention: 32 bh * 128 qtiles = 4096 wave-jobs, 4 waves/block
  attn_kernel<<<4096 / 4, 128, 0, stream>>>(hws);
  // 4) output projection: (4096/256) * (1024/64) = 16*16 = 256 blocks
  outproj_kernel<<<256, 128, 0, stream>>>(hws, out);
}
